// GGNN_60636348285345
// MI455X (gfx1250) — compile-verified, matrix operand order checked by reading, not run on hardware
//
#include <hip/hip_runtime.h>
#include <hip/hip_bf16.h>

// ---------------- problem constants ----------------
constexpr int N_NODES = 32768;
constexpr int N_EDGES = 524288;
constexpr int IN_DIM  = 128;
constexpr int D       = 256;     // OUT_DIM
constexpr int N_ETYPES = 4;
constexpr int N_STEPS = 8;
constexpr int BATCH   = 64;
constexpr int NPG     = N_NODES / BATCH;  // 512

typedef __attribute__((ext_vector_type(16))) __bf16 v16bf;
typedef __attribute__((ext_vector_type(8)))  __bf16 v8bf;
typedef __attribute__((ext_vector_type(8)))  float  v8f;

// Load a 16x32 bf16 WMMA operand tile (A or B identically) from a row-major
// [rows, ld] bf16 matrix. Lane l holds row (l&15); K-chunks at
// k + (l>>4)*8 and k + 16 + (l>>4)*8  (two 16-byte loads).
__device__ __forceinline__ v16bf load_tile_bf16(const __bf16* base, int ld,
                                                int row, int k, int lane) {
    const __bf16* p = base + (size_t)row * ld + k + ((lane >> 4) << 3);
    v8bf lo = *(const v8bf*)(p);
    v8bf hi = *(const v8bf*)(p + 16);
    return __builtin_shufflevector(lo, hi, 0, 1, 2, 3, 4, 5, 6, 7,
                                           8, 9, 10, 11, 12, 13, 14, 15);
}

__device__ __forceinline__ float sigmoid_f(float x) {
    return 1.0f / (1.0f + __expf(-x));
}
__device__ __forceinline__ float tanh_f(float x) {
    return 1.0f - 2.0f / (1.0f + __expf(2.0f * x));
}

// ---------------- init / conversion kernels ----------------
__global__ void init_h_kernel(const float* __restrict__ feat,
                              float* __restrict__ h,
                              __bf16* __restrict__ h_bf) {
    int idx = blockIdx.x * blockDim.x + threadIdx.x;   // N*D threads
    int n = idx >> 8, c = idx & 255;
    float v = (c < IN_DIM) ? feat[n * IN_DIM + c] : 0.0f;
    h[idx] = v;
    h_bf[idx] = (__bf16)v;
}

__global__ void cvt_bf16_kernel(const float* __restrict__ src,
                                __bf16* __restrict__ dst, int n) {
    int i = blockIdx.x * blockDim.x + threadIdx.x;
    if (i < n) dst[i] = (__bf16)src[i];
}

__global__ void zero_f32_kernel(float* __restrict__ p, int n) {
    int i = blockIdx.x * blockDim.x + threadIdx.x;
    if (i < n) p[i] = 0.0f;
}

// ---------------- per-etype transform: Ht[t] = h @ Wt[t]^T + bt[t] ---------
// One wave -> 16 rows x 64 cols (4 WMMA tiles sharing the A operand).
// grid: 32768 waves = 4 etypes * 2048 mtiles * 4 ngroups ; blockDim = 128.
__global__ __launch_bounds__(128)
void etype_gemm_kernel(const __bf16* __restrict__ h_bf,
                       const __bf16* __restrict__ Wt_bf,
                       const float*  __restrict__ bt,
                       float* __restrict__ Ht) {
    int lane = threadIdx.x & 31;
    int wid  = blockIdx.x * 4 + (threadIdx.x >> 5);
    int ng = wid & 3;
    int mt = (wid >> 2) & 2047;
    int t  = wid >> 13;
    int m0 = mt << 4;
    int n0 = ng << 6;

    const __bf16* A = h_bf;                                // rows = nodes
    const __bf16* W = Wt_bf + (size_t)t * D * D;           // rows = out-features

    v8f acc0 = {}, acc1 = {}, acc2 = {}, acc3 = {};
    int arow = m0 + (lane & 15);
    int brow = n0 + (lane & 15);
#pragma unroll
    for (int k = 0; k < D; k += 32) {
        v16bf a = load_tile_bf16(A, D, arow, k, lane);
        v16bf b0 = load_tile_bf16(W, D, brow,      k, lane);
        v16bf b1 = load_tile_bf16(W, D, brow + 16, k, lane);
        v16bf b2 = load_tile_bf16(W, D, brow + 32, k, lane);
        v16bf b3 = load_tile_bf16(W, D, brow + 48, k, lane);
        acc0 = __builtin_amdgcn_wmma_f32_16x16x32_bf16(false, a, false, b0, (short)0, acc0, false, false);
        acc1 = __builtin_amdgcn_wmma_f32_16x16x32_bf16(false, a, false, b1, (short)0, acc1, false, false);
        acc2 = __builtin_amdgcn_wmma_f32_16x16x32_bf16(false, a, false, b2, (short)0, acc2, false, false);
        acc3 = __builtin_amdgcn_wmma_f32_16x16x32_bf16(false, a, false, b3, (short)0, acc3, false, false);
    }

    // C layout: col = lane&15 (+tile offset), row = m0 + (lane>>4)*8 + i
    int col = lane & 15;
    int rbase = m0 + ((lane >> 4) << 3);
    float* out = Ht + (size_t)t * N_NODES * D;
    v8f accs[4] = {acc0, acc1, acc2, acc3};
#pragma unroll
    for (int j = 0; j < 4; ++j) {
        int c = n0 + j * 16 + col;
        float bias = bt[t * D + c];
#pragma unroll
        for (int i = 0; i < 8; ++i) {
            out[(size_t)(rbase + i) * D + c] = accs[j][i] + bias;
        }
    }
}

// ---------------- edge scatter: a[dst] += Ht[etype, src] -------------------
// 64 threads per edge, 4 channels each (float4 load + 4 hw fp32 atomics).
__global__ __launch_bounds__(256)
void edge_scatter_kernel(const float* __restrict__ Ht,
                         const int* __restrict__ src,
                         const int* __restrict__ dst,
                         const int* __restrict__ et,
                         float* __restrict__ a) {
    int tid = blockIdx.x * blockDim.x + threadIdx.x;
    int e = tid >> 6;
    int c = (tid & 63) << 2;
    int s = src[e], d = dst[e], t = et[e];
    const float4 v = *(const float4*)(Ht + ((size_t)t * N_NODES + s) * D + c);
    float* ap = a + (size_t)d * D + c;
    unsafeAtomicAdd(ap + 0, v.x);
    unsafeAtomicAdd(ap + 1, v.y);
    unsafeAtomicAdd(ap + 2, v.z);
    unsafeAtomicAdd(ap + 3, v.w);
}

// ---------------- fused GRU cell -------------------------------------------
// One wave -> 16 nodes x 16 output channels. Six WMMA accumulator chains
// (i_r,i_z,i_n from a ; h_r,h_z,h_n from h), then elementwise GRU.
// h (f32) updated in place (each element owned by one wave); the bf16 mirror
// is ping-pong buffered (hbf_in read by all col-tile waves -> must not alias
// the hbf_out being written).
// grid: 32768 waves = 2048 mtiles * 16 ctiles ; blockDim = 128.
__global__ __launch_bounds__(128)
void gru_fused_kernel(const __bf16* __restrict__ a_bf,
                      const __bf16* __restrict__ hbf_in,
                      const __bf16* __restrict__ wih_bf,   // [3D, D]
                      const __bf16* __restrict__ whh_bf,   // [3D, D]
                      const float*  __restrict__ b_ih,     // [3D]
                      const float*  __restrict__ b_hh,     // [3D]
                      float* __restrict__ h,
                      __bf16* __restrict__ hbf_out) {
    int lane = threadIdx.x & 31;
    int wid  = blockIdx.x * 4 + (threadIdx.x >> 5);
    int ct = wid & 15;
    int mt = wid >> 4;
    int m0 = mt << 4;
    int n0 = ct << 4;

    v8f air = {}, aiz = {}, ain = {};
    v8f ahr = {}, ahz = {}, ahn = {};
    int arow = m0 + (lane & 15);
    int wrow = n0 + (lane & 15);
#pragma unroll
    for (int k = 0; k < D; k += 32) {
        v16bf Aa = load_tile_bf16(a_bf, D, arow, k, lane);
        v16bf Ah = load_tile_bf16(hbf_in, D, arow, k, lane);
        v16bf Bir = load_tile_bf16(wih_bf, D, wrow,          k, lane);
        v16bf Biz = load_tile_bf16(wih_bf, D, wrow + D,      k, lane);
        v16bf Bin = load_tile_bf16(wih_bf, D, wrow + 2 * D,  k, lane);
        v16bf Bhr = load_tile_bf16(whh_bf, D, wrow,          k, lane);
        v16bf Bhz = load_tile_bf16(whh_bf, D, wrow + D,      k, lane);
        v16bf Bhn = load_tile_bf16(whh_bf, D, wrow + 2 * D,  k, lane);
        air = __builtin_amdgcn_wmma_f32_16x16x32_bf16(false, Aa, false, Bir, (short)0, air, false, false);
        aiz = __builtin_amdgcn_wmma_f32_16x16x32_bf16(false, Aa, false, Biz, (short)0, aiz, false, false);
        ain = __builtin_amdgcn_wmma_f32_16x16x32_bf16(false, Aa, false, Bin, (short)0, ain, false, false);
        ahr = __builtin_amdgcn_wmma_f32_16x16x32_bf16(false, Ah, false, Bhr, (short)0, ahr, false, false);
        ahz = __builtin_amdgcn_wmma_f32_16x16x32_bf16(false, Ah, false, Bhz, (short)0, ahz, false, false);
        ahn = __builtin_amdgcn_wmma_f32_16x16x32_bf16(false, Ah, false, Bhn, (short)0, ahn, false, false);
    }

    int col = n0 + (lane & 15);
    int rbase = m0 + ((lane >> 4) << 3);
    float bir = b_ih[col], biz = b_ih[D + col], bin = b_ih[2 * D + col];
    float bhr = b_hh[col], bhz = b_hh[D + col], bhn = b_hh[2 * D + col];
#pragma unroll
    for (int i = 0; i < 8; ++i) {
        size_t idx = (size_t)(rbase + i) * D + col;
        float r = sigmoid_f((air[i] + bir) + (ahr[i] + bhr));
        float z = sigmoid_f((aiz[i] + biz) + (ahz[i] + bhz));
        float n = tanh_f((ain[i] + bin) + r * (ahn[i] + bhn));
        float hnew = (1.0f - z) * n + z * h[idx];
        h[idx] = hnew;
        hbf_out[idx] = (__bf16)hnew;
    }
}

// ---------------- readout: per-graph sum -> linear -> sigmoid --------------
__global__ __launch_bounds__(256)
void readout_kernel(const float* __restrict__ h,
                    const float* __restrict__ w_cls,
                    const float* __restrict__ b_cls,
                    float* __restrict__ out) {
    __shared__ float red[256];
    int b = blockIdx.x;
    int d = threadIdx.x;
    const float* hp = h + (size_t)b * NPG * D + d;
    float s = 0.0f;
    for (int i = 0; i < NPG; ++i) s += hp[(size_t)i * D];
    red[d] = s * w_cls[d];
    __syncthreads();
    for (int off = 128; off > 0; off >>= 1) {
        if (d < off) red[d] += red[d + off];
        __syncthreads();
    }
    if (d == 0) out[b] = sigmoid_f(red[0] + b_cls[0]);
}

// ---------------- host orchestration ---------------------------------------
extern "C" void kernel_launch(void* const* d_in, const int* in_sizes, int n_in,
                              void* d_out, int out_size, void* d_ws, size_t ws_size,
                              hipStream_t stream) {
    const float* features = (const float*)d_in[0];
    const int*   src      = (const int*)d_in[1];
    const int*   dst      = (const int*)d_in[2];
    const int*   etypes   = (const int*)d_in[3];
    const float* Wt       = (const float*)d_in[4];
    const float* bt       = (const float*)d_in[5];
    const float* w_ih     = (const float*)d_in[6];
    const float* w_hh     = (const float*)d_in[7];
    const float* b_ih     = (const float*)d_in[8];
    const float* b_hh     = (const float*)d_in[9];
    const float* w_cls    = (const float*)d_in[10];
    const float* b_cls    = (const float*)d_in[11];
    float* out = (float*)d_out;

    // workspace layout
    char* ws = (char*)d_ws;
    const size_t ND = (size_t)N_NODES * D;
    float*  h     = (float*)(ws);                              // N*D f32
    float*  Ht    = (float*)(ws + ND * 4);                     // 4*N*D f32
    float*  a     = (float*)(ws + ND * 4 + (size_t)N_ETYPES * ND * 4);
    __bf16* h_bf0 = (__bf16*)((char*)a + ND * 4);              // ping
    __bf16* h_bf1 = (__bf16*)((char*)h_bf0 + ND * 2);          // pong
    __bf16* a_bf  = (__bf16*)((char*)h_bf1 + ND * 2);
    __bf16* Wt_bf  = (__bf16*)((char*)a_bf + ND * 2);          // 4*D*D
    __bf16* wih_bf = (__bf16*)((char*)Wt_bf + (size_t)N_ETYPES * D * D * 2);
    __bf16* whh_bf = (__bf16*)((char*)wih_bf + (size_t)3 * D * D * 2);

    const int nWt  = N_ETYPES * D * D;   // 262144
    const int nWih = 3 * D * D;          // 196608

    // one-time (per call) weight conversions + h init (f32 + bf16 mirror)
    cvt_bf16_kernel<<<(nWt + 255) / 256, 256, 0, stream>>>(Wt, Wt_bf, nWt);
    cvt_bf16_kernel<<<(nWih + 255) / 256, 256, 0, stream>>>(w_ih, wih_bf, nWih);
    cvt_bf16_kernel<<<(nWih + 255) / 256, 256, 0, stream>>>(w_hh, whh_bf, nWih);
    init_h_kernel<<<(int)(ND / 256), 256, 0, stream>>>(features, h, h_bf0);

    const int elem_blocks = (int)(ND / 256);          // 32768
    const int gemm_blocks = 8192;                     // 32768 waves / 4
    const int edge_blocks = N_EDGES * 64 / 256;       // 131072
    const int gru_blocks  = 8192;                     // 32768 waves / 4

    for (int step = 0; step < N_STEPS; ++step) {
        const __bf16* hbf_in = (step & 1) ? h_bf1 : h_bf0;
        __bf16* hbf_out      = (step & 1) ? h_bf0 : h_bf1;
        etype_gemm_kernel<<<gemm_blocks, 128, 0, stream>>>(hbf_in, Wt_bf, bt, Ht);
        zero_f32_kernel<<<elem_blocks, 256, 0, stream>>>(a, (int)ND);
        edge_scatter_kernel<<<edge_blocks, 256, 0, stream>>>(Ht, src, dst, etypes, a);
        cvt_bf16_kernel<<<elem_blocks, 256, 0, stream>>>(a, a_bf, (int)ND);
        gru_fused_kernel<<<gru_blocks, 128, 0, stream>>>(a_bf, hbf_in, wih_bf, whh_bf,
                                                         b_ih, b_hh, h, hbf_out);
    }

    readout_kernel<<<BATCH, 256, 0, stream>>>(h, w_cls, b_cls, out);
}
